// Model_19868518711349
// MI455X (gfx1250) — matrix operand order repaired and checked by reading, not compiled, not run on hardware
//
#include <hip/hip_runtime.h>
#include <hip/hip_bf16.h>
#include <math.h>

// ---------------------------------------------------------------------------
// Non-stationary Transformer forward for MI455X (gfx1250, wave32, WMMA).
// All GEMMs run through one bf16 WMMA kernel (v_wmma_f32_16x16x32_bf16),
// 128x128 block tile, BK=32, 8 waves, each wave 32x64 (2x4 WMMA tiles).
// Staging: float4 global loads -> packed bf16 cvt -> b64 LDS stores, with
// register double-buffering so global loads overlap the WMMA chain.
// ---------------------------------------------------------------------------

typedef __attribute__((ext_vector_type(16))) __bf16 v16bf;
typedef __attribute__((ext_vector_type(8)))  float  v8f;

union FragB16 { v16bf v; unsigned u[8]; };
union PackBF4 { __bf16 h[4]; uint2 u2; };

#define BM 128
#define BN 128
#define BK 32

// C[M,N] = act( A[M,K] * B(K,N) + bias ), batched via blockIdx.z -> (b,h).
// bT==1 reads B element (k,n) as Bsrc[n*ldb + k] (i.e. B = Bsrc^T).
__global__ __launch_bounds__(256) void k_gemm(
    const float* __restrict__ Ag, int lda,
    const float* __restrict__ Bg, int ldb, int bT,
    float* __restrict__ Cg, int ldc,
    int M, int N, int K,
    const float* __restrict__ bias, int relu,
    int H, long long sAb, long long sAh, long long sBb, long long sBh,
    long long sCb, long long sCh)
{
    __shared__ __bf16 lsA[BM * BK];      // [m][k]
    __shared__ __bf16 lsB[BN * BK];      // [n][k]  (B stored transposed)

    int z  = blockIdx.z;
    int bb = z / H, hh = z - bb * H;
    const float* A = Ag + (size_t)(bb * sAb + hh * sAh);
    const float* B = Bg + (size_t)(bb * sBb + hh * sBh);
    float*       C = Cg + (size_t)(bb * sCb + hh * sCh);

    int m0 = blockIdx.y * BM, n0 = blockIdx.x * BN;
    int tid  = threadIdx.x;
    int lane = tid & 31, wave = tid >> 5;
    int waveM = (wave & 3) * 32;     // 4 waves along M
    int waveN = (wave >> 2) * 64;    // 2 waves along N

    const bool alignedA = ((lda & 3) == 0) && ((((unsigned long long)A) & 15ull) == 0);
    const bool alignedB = ((ldb & 3) == 0) && ((((unsigned long long)B) & 15ull) == 0);

    v8f vzero = {0.f, 0.f, 0.f, 0.f, 0.f, 0.f, 0.f, 0.f};
    v8f acc[2][4];
    for (int i = 0; i < 2; ++i)
        for (int j = 0; j < 4; ++j) acc[i][j] = vzero;

    float ra[4][4];                  // staged A: 4 x float4 per thread
    float rb[4][4];                  // staged B: 4 x float4 per thread

    // ---- tile loaders: global -> registers (float4 granularity) ----
    auto loadA = [&](int k0) {
        #pragma unroll
        for (int i = 0; i < 4; ++i) {
            int idx4 = i * 256 + tid;            // float4 index in 128x32 tile
            int r = idx4 >> 3, c4 = (idx4 & 7) << 2;
            int gr = m0 + r, gc = k0 + c4;
            float v0 = 0.f, v1 = 0.f, v2 = 0.f, v3 = 0.f;
            if (gr < M) {
                const float* p = A + (size_t)gr * lda + gc;
                if (alignedA && gc + 3 < K) {
                    float4 t = *(const float4*)p;
                    v0 = t.x; v1 = t.y; v2 = t.z; v3 = t.w;
                } else {
                    if (gc     < K) v0 = p[0];
                    if (gc + 1 < K) v1 = p[1];
                    if (gc + 2 < K) v2 = p[2];
                    if (gc + 3 < K) v3 = p[3];
                }
            }
            ra[i][0] = v0; ra[i][1] = v1; ra[i][2] = v2; ra[i][3] = v3;
        }
    };
    auto loadB = [&](int k0) {
        if (bT) {
            #pragma unroll
            for (int i = 0; i < 4; ++i) {
                int idx4 = i * 256 + tid;        // [n][k] tile, float4 along k
                int n = idx4 >> 3, k4 = (idx4 & 7) << 2;
                int gn = n0 + n, gk = k0 + k4;
                float v0 = 0.f, v1 = 0.f, v2 = 0.f, v3 = 0.f;
                if (gn < N) {
                    const float* p = B + (size_t)gn * ldb + gk;
                    if (alignedB && gk + 3 < K) {
                        float4 t = *(const float4*)p;
                        v0 = t.x; v1 = t.y; v2 = t.z; v3 = t.w;
                    } else {
                        if (gk     < K) v0 = p[0];
                        if (gk + 1 < K) v1 = p[1];
                        if (gk + 2 < K) v2 = p[2];
                        if (gk + 3 < K) v3 = p[3];
                    }
                }
                rb[i][0] = v0; rb[i][1] = v1; rb[i][2] = v2; rb[i][3] = v3;
            }
        } else {
            #pragma unroll
            for (int i = 0; i < 4; ++i) {
                int idx4 = i * 256 + tid;        // float4 along n (coalesced)
                int kk = idx4 >> 5, n4 = (idx4 & 31) << 2;
                int gk = k0 + kk, gn = n0 + n4;
                float v0 = 0.f, v1 = 0.f, v2 = 0.f, v3 = 0.f;
                if (gk < K) {
                    const float* p = B + (size_t)gk * ldb + gn;
                    if (alignedB && gn + 3 < N) {
                        float4 t = *(const float4*)p;
                        v0 = t.x; v1 = t.y; v2 = t.z; v3 = t.w;
                    } else {
                        if (gn     < N) v0 = p[0];
                        if (gn + 1 < N) v1 = p[1];
                        if (gn + 2 < N) v2 = p[2];
                        if (gn + 3 < N) v3 = p[3];
                    }
                }
                rb[i][0] = v0; rb[i][1] = v1; rb[i][2] = v2; rb[i][3] = v3;
            }
        }
    };
    // ---- registers -> LDS (packed bf16, b64 stores) ----
    auto stageStore = [&]() {
        #pragma unroll
        for (int i = 0; i < 4; ++i) {
            int idx4 = i * 256 + tid;
            PackBF4 pk;
            pk.h[0] = (__bf16)ra[i][0]; pk.h[1] = (__bf16)ra[i][1];
            pk.h[2] = (__bf16)ra[i][2]; pk.h[3] = (__bf16)ra[i][3];
            ((uint2*)lsA)[idx4] = pk.u2;
        }
        if (bT) {
            #pragma unroll
            for (int i = 0; i < 4; ++i) {
                int idx4 = i * 256 + tid;
                PackBF4 pk;
                pk.h[0] = (__bf16)rb[i][0]; pk.h[1] = (__bf16)rb[i][1];
                pk.h[2] = (__bf16)rb[i][2]; pk.h[3] = (__bf16)rb[i][3];
                ((uint2*)lsB)[idx4] = pk.u2;
            }
        } else {
            #pragma unroll
            for (int i = 0; i < 4; ++i) {
                int idx4 = i * 256 + tid;
                int kk = idx4 >> 5, n4 = (idx4 & 31) << 2;
                #pragma unroll
                for (int e = 0; e < 4; ++e)
                    lsB[(n4 + e) * BK + kk] = (__bf16)rb[i][e];
            }
        }
    };

    // ---- pipelined main loop: load k+BK while computing k ----
    loadA(0); loadB(0);
    for (int k0 = 0; k0 < K; k0 += BK) {
        stageStore();
        __syncthreads();
        if (k0 + BK < K) { loadA(k0 + BK); loadB(k0 + BK); }  // in flight during WMMA

        FragB16 af[2], bfr[4];
        int rsel  = lane & 15;
        int ahalf = (lane < 16) ? 0 : 8;    // A: lanes16-31 hold K=8..15,24..31
        #pragma unroll
        for (int mi = 0; mi < 2; ++mi) {
            int row = waveM + mi * 16 + rsel;
            const __bf16* base = &lsA[row * BK];
            #pragma unroll
            for (int p = 0; p < 8; ++p) {
                int kIdx = (p < 4) ? (ahalf + 2 * p) : (16 + ahalf + 2 * (p - 4));
                af[mi].u[p] = *(const unsigned*)(base + kIdx);
            }
        }
        int bhalf = (lane < 16) ? 0 : 8;    // B: lanes16-31 hold K=8..15,24..31
        #pragma unroll
        for (int ni = 0; ni < 4; ++ni) {
            int col = waveN + ni * 16 + rsel;
            const __bf16* base = &lsB[col * BK];
            #pragma unroll
            for (int p = 0; p < 8; ++p) {
                int kIdx = (p < 4) ? (bhalf + 2 * p) : (16 + bhalf + 2 * (p - 4));
                bfr[ni].u[p] = *(const unsigned*)(base + kIdx);
            }
        }

        #pragma unroll
        for (int mi = 0; mi < 2; ++mi)
            #pragma unroll
            for (int ni = 0; ni < 4; ++ni)
                acc[mi][ni] = __builtin_amdgcn_wmma_f32_16x16x32_bf16(
                    false, af[mi].v, false, bfr[ni].v,
                    (short)0, acc[mi][ni], false, false);
        __syncthreads();
    }

    // ---- epilogue: C/D f32 layout: vgpr r, lane l -> (M = r or r+8, N = l&15)
    for (int mi = 0; mi < 2; ++mi)
        for (int ni = 0; ni < 4; ++ni) {
            int col = n0 + waveN + ni * 16 + (lane & 15);
            if (col >= N) continue;
            float badd = bias ? bias[col] : 0.f;
            #pragma unroll
            for (int r = 0; r < 8; ++r) {
                int row = m0 + waveM + mi * 16 + ((lane < 16) ? r : r + 8);
                if (row < M) {
                    float v = acc[mi][ni][r] + badd;
                    if (relu) v = fmaxf(v, 0.f);
                    C[(size_t)row * ldc + col] = v;
                }
            }
        }
}

// ---------------------------------------------------------------------------
// per-(b,c) mean/std over SEQ=512
__global__ __launch_bounds__(256) void k_stats(const float* __restrict__ x,
                                               float* __restrict__ mean,
                                               float* __restrict__ stdv)
{
    __shared__ float red[256];
    int bc = blockIdx.x;                 // b*21+c
    int b = bc / 21, c = bc % 21;
    int tid = threadIdx.x;
    float s = 0.f, sq = 0.f;
    for (int i = tid; i < 512; i += 256) {
        float v = x[((size_t)b * 512 + i) * 21 + c];
        s += v; sq += v * v;
    }
    red[tid] = s; __syncthreads();
    for (int t = 128; t > 0; t >>= 1) { if (tid < t) red[tid] += red[tid + t]; __syncthreads(); }
    s = red[0]; __syncthreads();
    red[tid] = sq; __syncthreads();
    for (int t = 128; t > 0; t >>= 1) { if (tid < t) red[tid] += red[tid + t]; __syncthreads(); }
    sq = red[0];
    if (tid == 0) {
        float m = s / 512.f;
        mean[bc] = m;
        stdv[bc] = sqrtf(fmaxf(sq / 512.f - m * m, 0.f) + 1e-5f);
    }
}

__global__ void k_norm(const float* __restrict__ x, const float* __restrict__ mean,
                       const float* __restrict__ stdv, float* __restrict__ xn, int total)
{
    int i = blockIdx.x * 256 + threadIdx.x;
    if (i >= total) return;
    int c = i % 21;
    int b = i / (512 * 21);
    xn[i] = (x[i] - mean[b * 21 + c]) / stdv[b * 21 + c];
}

// x_dec_new = concat(x_enc_n[:, -256:, :], zeros(B,256,21))
__global__ void k_decin(const float* __restrict__ xn, float* __restrict__ dc, int total)
{
    int i = blockIdx.x * 256 + threadIdx.x;
    if (i >= total) return;
    int c = i % 21;
    int t = (i / 21) % 512;
    int b = i / (512 * 21);
    dc[i] = (t < 256) ? xn[((size_t)b * 512 + 256 + t) * 21 + c] : 0.f;
}

// projector series conv: mlpin[b, 0:21] = circconv(x_raw^T)[b,e]+cb,
//                        mlpin[b,21:42] = stats[b,e]
__global__ __launch_bounds__(256) void k_sconv(const float* __restrict__ xraw,
                                               const float* __restrict__ cw,
                                               const float* __restrict__ cb,
                                               const float* __restrict__ stats,
                                               float* __restrict__ mlpin)
{
    __shared__ float red[256];
    int be = blockIdx.x;                // b*21+e
    int b = be / 21, e = be % 21;
    int tid = threadIdx.x;
    float s = 0.f;
    for (int j = tid; j < 3 * 512; j += 256) {
        int k = j >> 9, sidx = j & 511;
        int ee = (e + k - 1 + 21) % 21;
        s += xraw[((size_t)b * 512 + sidx) * 21 + ee] * cw[j];
    }
    red[tid] = s; __syncthreads();
    for (int t = 128; t > 0; t >>= 1) { if (tid < t) red[tid] += red[tid + t]; __syncthreads(); }
    if (tid == 0) {
        mlpin[b * 42 + e]      = red[0] + cb[0];
        mlpin[b * 42 + 21 + e] = stats[b * 21 + e];
    }
}

__global__ void k_exp(const float* __restrict__ a, float* __restrict__ o, int n)
{
    int i = blockIdx.x * 64 + threadIdx.x;
    if (i < n) o[i] = __expf(a[i]);
}

// token circular-conv embedding + sinusoidal PEs (month, hour, position)
__global__ __launch_bounds__(256) void k_embed(const float* __restrict__ xin,
                                               const int* __restrict__ marks,
                                               const float* __restrict__ cw,  // (3,21,512)
                                               float* __restrict__ out)
{
    __shared__ float xs[63];
    int bl = blockIdx.x;                // b*512 + l
    int b = bl >> 9, l = bl & 511;
    int tid = threadIdx.x;
    if (tid < 63) {
        int k = tid / 21, c = tid % 21;
        int ll = (l + k - 1 + 512) & 511;
        xs[tid] = xin[((size_t)b * 512 + ll) * 21 + c];
    }
    __syncthreads();
    int m0 = marks[((size_t)b * 512 + l) * 2 + 0];
    int m1 = marks[((size_t)b * 512 + l) * 2 + 1];
    const float kln = 9.210340372f / 512.f;   // ln(10000)/d_model
    for (int d = tid; d < 512; d += 256) {
        float acc = 0.f;
        #pragma unroll 9
        for (int j = 0; j < 63; ++j) acc += xs[j] * cw[j * 512 + d];
        float w = __expf(-(float)(d & ~1) * kln);
        float pe;
        if (d & 1) pe = __cosf(m0 * w) + __cosf(m1 * w) + __cosf(l * w);
        else       pe = __sinf(m0 * w) + __sinf(m1 * w) + __sinf(l * w);
        out[((size_t)b * 512 + l) * 512 + d] = acc + pe;
    }
}

// masked de-stationary softmax over rows of the score matrix.
// scores layout: [(b*H+h), Lq, S]; v = scale*(score*tau[b] + delta[b,s]); causal mask.
__global__ __launch_bounds__(256) void k_softmax(float* __restrict__ sc,
                                                 const float* __restrict__ tau,
                                                 const float* __restrict__ delta,
                                                 int Lq, int S, int H,
                                                 int causal, float scale)
{
    __shared__ float red[256];
    int row = blockIdx.x;               // bh*Lq + l
    int l  = row % Lq;
    int bh = row / Lq;
    int b  = bh / H;
    float* p = sc + (size_t)row * S;
    float t = tau[b];
    int tid = threadIdx.x;

    float mx = -3.4e38f;
    for (int i = tid; i < S; i += 256) {
        float v = p[i] * t;
        if (delta) v += delta[b * S + i];
        v *= scale;
        if (causal && i > l) v = -3.4e38f;
        p[i] = v;
        mx = fmaxf(mx, v);
    }
    red[tid] = mx; __syncthreads();
    for (int s2 = 128; s2 > 0; s2 >>= 1) { if (tid < s2) red[tid] = fmaxf(red[tid], red[tid + s2]); __syncthreads(); }
    mx = red[0]; __syncthreads();

    float sum = 0.f;
    for (int i = tid; i < S; i += 256) {
        float e = __expf(p[i] - mx);
        p[i] = e; sum += e;
    }
    red[tid] = sum; __syncthreads();
    for (int s2 = 128; s2 > 0; s2 >>= 1) { if (tid < s2) red[tid] += red[tid + s2]; __syncthreads(); }
    float inv = 1.0f / red[0];
    for (int i = tid; i < S; i += 256) p[i] *= inv;
}

// y = LN(x (+ r)) * g + b over rows of 512
__global__ __launch_bounds__(256) void k_ln(const float* __restrict__ x,
                                            const float* __restrict__ r,
                                            float* __restrict__ y,
                                            const float* __restrict__ g,
                                            const float* __restrict__ b)
{
    __shared__ float red[256];
    size_t row = blockIdx.x;
    int tid = threadIdx.x;
    const float* xp = x + row * 512;
    float v0 = xp[tid]       + (r ? r[row * 512 + tid]       : 0.f);
    float v1 = xp[tid + 256] + (r ? r[row * 512 + tid + 256] : 0.f);
    red[tid] = v0 + v1; __syncthreads();
    for (int t = 128; t > 0; t >>= 1) { if (tid < t) red[tid] += red[tid + t]; __syncthreads(); }
    float m = red[0] / 512.f; __syncthreads();
    float d0 = v0 - m, d1 = v1 - m;
    red[tid] = d0 * d0 + d1 * d1; __syncthreads();
    for (int t = 128; t > 0; t >>= 1) { if (tid < t) red[tid] += red[tid + t]; __syncthreads(); }
    float rinv = rsqrtf(red[0] / 512.f + 1e-5f);
    y[row * 512 + tid]       = d0 * rinv * g[tid]       + b[tid];
    y[row * 512 + tid + 256] = d1 * rinv * g[tid + 256] + b[tid + 256];
}

// out[b,t,c] = proj[b,256+t,c]*std[b,c] + mean[b,c]
__global__ void k_final(const float* __restrict__ dp, const float* __restrict__ mean,
                        const float* __restrict__ stdv, float* __restrict__ out, int total)
{
    int i = blockIdx.x * 256 + threadIdx.x;
    if (i >= total) return;
    int c = i % 21;
    int t = (i / 21) % 256;
    int b = i / (256 * 21);
    out[i] = dp[((size_t)b * 512 + 256 + t) * 21 + c] * stdv[b * 21 + c] + mean[b * 21 + c];
}

// ---------------------------------------------------------------------------

static inline void gemm(hipStream_t st, const float* A, int lda,
                        const float* B, int ldb, int bT,
                        float* C, int ldc, int M, int N, int K,
                        const float* bias, int relu,
                        int batch = 1, int H = 1,
                        long long sAb = 0, long long sAh = 0,
                        long long sBb = 0, long long sBh = 0,
                        long long sCb = 0, long long sCh = 0)
{
    dim3 g((N + BN - 1) / BN, (M + BM - 1) / BM, batch);
    k_gemm<<<g, 256, 0, st>>>(A, lda, B, ldb, bT, C, ldc, M, N, K,
                              bias, relu, H, sAb, sAh, sBb, sBh, sCb, sCh);
}

struct AttnW { const float *Wk, *Wo, *Wq, *Wv, *bk, *bo, *bq, *bv; };

extern "C" void kernel_launch(void* const* d_in, const int* in_sizes, int n_in,
                              void* d_out, int out_size, void* d_ws, size_t ws_size,
                              hipStream_t stream)
{
    (void)in_sizes; (void)n_in; (void)out_size; (void)ws_size;
    auto F = [&](int i) { return (const float*)d_in[i]; };

    // --- inputs (jax pytree: dict keys flattened in sorted order) ------------
    const float* x_enc  = F(0);
    const int*   mk_enc = (const int*)d_in[1];
    const float* x_dec  = F(2); (void)x_dec;     // only zeros/labels used
    const int*   mk_dec = (const int*)d_in[3];

    const float* dec_conv_w = F(4);
    // dec_layers[0]: cross, ffn, ln1..ln3, self
    AttnW cross = { F(5), F(6), F(7), F(8), F(9), F(10), F(11), F(12) };
    const float *dfb1 = F(13), *dfb2 = F(14), *dfw1 = F(15), *dfw2 = F(16);
    const float *dln1b = F(17), *dln1g = F(18);
    const float *dln2b = F(19), *dln2g = F(20);
    const float *dln3b = F(21), *dln3g = F(22);
    AttnW dself = { F(23), F(24), F(25), F(26), F(27), F(28), F(29), F(30) };
    const float *dnb = F(31), *dng = F(32);
    // delta projector: b0,b1,conv_b,conv_w,w0,w1,w2
    const float *del_b0 = F(33), *del_b1 = F(34), *del_cb = F(35), *del_cw = F(36);
    const float *del_w0 = F(37), *del_w1 = F(38), *del_w2 = F(39);
    const float* enc_conv_w = F(40);
    AttnW encA[2] = { { F(41), F(42), F(43), F(44), F(45), F(46), F(47), F(48) },
                      { F(57), F(58), F(59), F(60), F(61), F(62), F(63), F(64) } };
    const float *efb1[2] = { F(49), F(65) }, *efb2[2] = { F(50), F(66) };
    const float *efw1[2] = { F(51), F(67) }, *efw2[2] = { F(52), F(68) };
    const float *eln1b[2] = { F(53), F(69) }, *eln1g[2] = { F(54), F(70) };
    const float *eln2b[2] = { F(55), F(71) }, *eln2g[2] = { F(56), F(72) };
    const float *enb = F(73), *eng = F(74);
    const float *proj_b = F(75), *proj_w = F(76);
    const float *tau_b0 = F(77), *tau_b1 = F(78), *tau_cb = F(79), *tau_cw = F(80);
    const float *tau_w0 = F(81), *tau_w1 = F(82), *tau_w2 = F(83);

    // --- workspace layout (floats) ------------------------------------------
    float* W = (float*)d_ws;
    size_t off = 0;
    auto ALLOC = [&](size_t n) { float* p = W + off; off += (n + 255) & ~(size_t)255; return p; };
    const int Bm = 16, L = 512, D = 512, H = 8, C = 21, DFF = 2048;
    const size_t ACT = (size_t)Bm * L * D;          // 4.19M floats
    float* mean   = ALLOC(Bm * C);
    float* stdv   = ALLOC(Bm * C);
    float* xn     = ALLOC((size_t)Bm * L * C);
    float* decin  = ALLOC((size_t)Bm * L * C);
    float* tin    = ALLOC(Bm * 42);
    float* din    = ALLOC(Bm * 42);
    float* h1     = ALLOC(Bm * 128);
    float* h2     = ALLOC(Bm * 128);
    float* traw   = ALLOC(Bm);
    float* tauv   = ALLOC(Bm);
    float* deltav = ALLOC((size_t)Bm * L);
    float* encx   = ALLOC(ACT);
    float* ENC    = ALLOC(ACT);
    float* DX     = ALLOC(ACT);
    float* G0 = ALLOC(ACT); float* G1 = ALLOC(ACT); float* G2 = ALLOC(ACT);
    float* G3 = ALLOC(ACT); float* G4 = ALLOC(ACT); float* G5 = ALLOC(ACT);
    float* SC = ALLOC((size_t)Bm * H * L * L);      // 33.6M floats
    float* FH = SC;                                  // FFN hidden reuses scores
    float* DP = ALLOC((size_t)Bm * L * C);

    const int MR = Bm * L;                  // 8192 rows
    const long long sA = (long long)L * D;  // per-b activation stride
    const long long sS = (long long)L * L;  // per-head score stride
    const float scale = 0.125f;             // 1/sqrt(64)

    // --- stats, normalization, decoder input --------------------------------
    k_stats<<<Bm * C, 256, 0, stream>>>(x_enc, mean, stdv);
    k_norm <<<(Bm * L * C + 255) / 256, 256, 0, stream>>>(x_enc, mean, stdv, xn, Bm * L * C);
    k_decin<<<(Bm * L * C + 255) / 256, 256, 0, stream>>>(xn, decin, Bm * L * C);

    // --- tau/delta projectors (inputs use RAW x_enc) ------------------------
    k_sconv<<<Bm * C, 256, 0, stream>>>(x_enc, tau_cw, tau_cb, stdv, tin);
    k_sconv<<<Bm * C, 256, 0, stream>>>(x_enc, del_cw, del_cb, mean, din);
    gemm(stream, tin, 42, tau_w0, 128, 0, h1, 128, Bm, 128, 42, tau_b0, 1);
    gemm(stream, h1, 128, tau_w1, 128, 0, h2, 128, Bm, 128, 128, tau_b1, 1);
    gemm(stream, h2, 128, tau_w2, 1, 0, traw, 1, Bm, 1, 128, nullptr, 0);
    k_exp<<<1, 64, 0, stream>>>(traw, tauv, Bm);
    gemm(stream, din, 42, del_w0, 128, 0, h1, 128, Bm, 128, 42, del_b0, 1);
    gemm(stream, h1, 128, del_w1, 128, 0, h2, 128, Bm, 128, 128, del_b1, 1);
    gemm(stream, h2, 128, del_w2, L, 0, deltav, L, Bm, L, 128, nullptr, 0);

    // --- encoder -------------------------------------------------------------
    k_embed<<<Bm * L, 256, 0, stream>>>(xn, mk_enc, enc_conv_w, encx);
    for (int li = 0; li < 2; ++li) {
        const AttnW& a = encA[li];
        gemm(stream, encx, D, a.Wq, D, 0, G0, D, MR, D, D, a.bq, 0);
        gemm(stream, encx, D, a.Wk, D, 0, G1, D, MR, D, D, a.bk, 0);
        gemm(stream, encx, D, a.Wv, D, 0, G2, D, MR, D, D, a.bv, 0);
        // scores[b,h] = Q_h K_h^T   (M=N=512, K=64, batch=B*H)
        gemm(stream, G0, D, G1, D, 1, SC, L, L, L, 64, nullptr, 0,
             Bm * H, H, sA, 64, sA, 64, (long long)H * sS, sS);
        k_softmax<<<Bm * H * L, 256, 0, stream>>>(SC, tauv, deltav, L, L, H, 0, scale);
        // attn = P V  (M=512, N=64, K=512)
        gemm(stream, SC, L, G2, D, 0, G3, D, L, 64, L, nullptr, 0,
             Bm * H, H, (long long)H * sS, sS, sA, 64, sA, 64);
        gemm(stream, G3, D, a.Wo, D, 0, G4, D, MR, D, D, a.bo, 0);
        k_ln<<<MR, 256, 0, stream>>>(encx, G4, G5, eln1g[li], eln1b[li]);
        gemm(stream, G5, D, efw1[li], DFF, 0, FH, DFF, MR, DFF, D, efb1[li], 1);
        gemm(stream, FH, DFF, efw2[li], D, 0, G3, D, MR, D, DFF, efb2[li], 0);
        k_ln<<<MR, 256, 0, stream>>>(G5, G3, encx, eln2g[li], eln2b[li]);
    }
    k_ln<<<MR, 256, 0, stream>>>(encx, nullptr, ENC, eng, enb);

    // --- decoder -------------------------------------------------------------
    k_embed<<<Bm * L, 256, 0, stream>>>(decin, mk_dec, dec_conv_w, DX);
    // self-attention (causal, tau only)
    gemm(stream, DX, D, dself.Wq, D, 0, G0, D, MR, D, D, dself.bq, 0);
    gemm(stream, DX, D, dself.Wk, D, 0, G1, D, MR, D, D, dself.bk, 0);
    gemm(stream, DX, D, dself.Wv, D, 0, G2, D, MR, D, D, dself.bv, 0);
    gemm(stream, G0, D, G1, D, 1, SC, L, L, L, 64, nullptr, 0,
         Bm * H, H, sA, 64, sA, 64, (long long)H * sS, sS);
    k_softmax<<<Bm * H * L, 256, 0, stream>>>(SC, tauv, nullptr, L, L, H, 1, scale);
    gemm(stream, SC, L, G2, D, 0, G3, D, L, 64, L, nullptr, 0,
         Bm * H, H, (long long)H * sS, sS, sA, 64, sA, 64);
    gemm(stream, G3, D, dself.Wo, D, 0, G4, D, MR, D, D, dself.bo, 0);
    k_ln<<<MR, 256, 0, stream>>>(DX, G4, DX, dln1g, dln1b);
    // cross-attention (tau + delta, keys/values from encoder)
    gemm(stream, DX,  D, cross.Wq, D, 0, G0, D, MR, D, D, cross.bq, 0);
    gemm(stream, ENC, D, cross.Wk, D, 0, G1, D, MR, D, D, cross.bk, 0);
    gemm(stream, ENC, D, cross.Wv, D, 0, G2, D, MR, D, D, cross.bv, 0);
    gemm(stream, G0, D, G1, D, 1, SC, L, L, L, 64, nullptr, 0,
         Bm * H, H, sA, 64, sA, 64, (long long)H * sS, sS);
    k_softmax<<<Bm * H * L, 256, 0, stream>>>(SC, tauv, deltav, L, L, H, 0, scale);
    gemm(stream, SC, L, G2, D, 0, G3, D, L, 64, L, nullptr, 0,
         Bm * H, H, (long long)H * sS, sS, sA, 64, sA, 64);
    gemm(stream, G3, D, cross.Wo, D, 0, G4, D, MR, D, D, cross.bo, 0);
    k_ln<<<MR, 256, 0, stream>>>(DX, G4, G5, dln2g, dln2b);
    // FFN + ln3 + final norm
    gemm(stream, G5, D, dfw1, DFF, 0, FH, DFF, MR, DFF, D, dfb1, 1);
    gemm(stream, FH, DFF, dfw2, D, 0, G3, D, MR, D, DFF, dfb2, 0);
    k_ln<<<MR, 256, 0, stream>>>(G5, G3, DX, dln3g, dln3b);
    k_ln<<<MR, 256, 0, stream>>>(DX, nullptr, G4, dng, dnb);

    // --- head + de-normalization --------------------------------------------
    gemm(stream, G4, D, proj_w, C, 0, DP, C, MR, C, D, proj_b, 0);
    k_final<<<(Bm * 256 * C + 255) / 256, 256, 0, stream>>>(DP, mean, stdv,
                                                            (float*)d_out, Bm * 256 * C);
}
